// BothGuideAttentionLayer_26972394619674
// MI455X (gfx1250) — matrix operand order repaired and checked by reading, not compiled, not run on hardware
//
#include <hip/hip_runtime.h>
#include <hip/hip_bf16.h>
#include <math.h>
#include <stdint.h>

typedef __bf16 bf16_t;
typedef __bf16 v16bf __attribute__((ext_vector_type(16)));
typedef float  v8f   __attribute__((ext_vector_type(8)));

#define TM 64      // block tile M
#define TN 128     // block tile N
#define TK 64      // staged K per round (2 WMMA substeps of 32)

// ---------------------------------------------------------------------------
// Stage `rows` x 64-half tile (row-major, leading dim ld) into LDS [row][k].
// Fast path: CDNA5 async global->LDS DMA (ASYNCcnt, no VGPR bounce).
// Edge path: zero-filled VGPR bounce + ds_store.
// ---------------------------------------------------------------------------
__device__ __forceinline__ void stage_tile(bf16_t* __restrict__ lds,
                                           const bf16_t* __restrict__ src,
                                           int rows, int ld, int rowLimit,
                                           int kBase, int K, int tid, int nThreads)
{
    const int nChunks = rows * 8;                 // 8 halves (16B) per chunk
    for (int c = tid; c < nChunks; c += nThreads) {
        int r  = c >> 3;
        int ko = (c & 7) * 8;
        bf16_t* dst = lds + r * TK + ko;
        if (r < rowLimit && (kBase + ko + 8) <= K) {
            unsigned lo = (unsigned)(uintptr_t)dst;          // LDS offset (low 32b of flat)
            unsigned long long ga =
                (unsigned long long)(uintptr_t)(src + (long long)r * ld + kBase + ko);
            asm volatile("global_load_async_to_lds_b128 %0, %1, off"
                         :: "v"(lo), "v"(ga) : "memory");
        } else {
            alignas(16) bf16_t tmp[8];
            for (int j = 0; j < 8; ++j)
                tmp[j] = (r < rowLimit && (kBase + ko + j) < K)
                           ? src[(long long)r * ld + kBase + ko + j] : (bf16_t)0.f;
            *(uint4*)dst = *(uint4*)tmp;
        }
    }
}

// ---------------------------------------------------------------------------
// Tiled bf16 WMMA GEMM: C = act( scale * (A @ B^T) + bias )
//   A: [batch, M, K] bf16 row-major
//   B: [batch, N, K] bf16 row-major (always N-major)
//   C: [batch, M, N] (outTrans=0) or [batch, N, M] (outTrans=1), f32 or bf16
// Block: 128 threads (4 waves), 64x128 C tile; each wave a 32x64 subtile.
// Double-buffered LDS; async DMA for tile t+1 overlaps WMMA on tile t.
// ---------------------------------------------------------------------------
__global__ __launch_bounds__(128) void gemm_bf16_wmma(
    const bf16_t* __restrict__ A, const bf16_t* __restrict__ Bm,
    const float* __restrict__ bias, void* __restrict__ Cout,
    int M, int N, int K, int lda, int ldb, int ldc,
    long long aBatch, long long bBatch, long long cBatch,
    int outBf16, int outTrans, int doRelu, float scale)
{
    __shared__ alignas(16) bf16_t sA[2][TM * TK];   // 2 x 8 KB
    __shared__ alignas(16) bf16_t sB[2][TN * TK];   // 2 x 16 KB

    const int tid  = threadIdx.x;
    const int lane = tid & 31;
    const int wv   = tid >> 5;
    const int h    = lane >> 4;
    const int l16  = lane & 15;
    const int m0 = blockIdx.y * TM;
    const int n0 = blockIdx.x * TN;
    const int wm = (wv >> 1) * 32;   // wave row offset (0/32)
    const int wn = (wv & 1) * 64;    // wave col offset (0/64)

    const bf16_t* Ab = A  + (long long)blockIdx.z * aBatch + (long long)m0 * lda;
    const bf16_t* Bb = Bm + (long long)blockIdx.z * bBatch + (long long)n0 * ldb;
    const int mLim = (M - m0) < TM ? (M - m0) : TM;
    const int nLim = (N - n0) < TN ? (N - n0) : TN;

    v8f acc[2][4] = {};

    const int nT = (K + TK - 1) / TK;

    // prologue: stage tile 0
    stage_tile(sA[0], Ab, TM, lda, mLim, 0, K, tid, 128);
    stage_tile(sB[0], Bb, TN, ldb, nLim, 0, K, tid, 128);
    asm volatile("s_wait_asynccnt 0x0" ::: "memory");
    __syncthreads();

    for (int t = 0; t < nT; ++t) {
        const int cur = t & 1;
        // issue async DMA for tile t+1 into the alternate buffer
        if (t + 1 < nT) {
            stage_tile(sA[cur ^ 1], Ab, TM, lda, mLim, (t + 1) * TK, K, tid, 128);
            stage_tile(sB[cur ^ 1], Bb, TN, ldb, nLim, (t + 1) * TK, K, tid, 128);
            // warm L2 for tile t+2 (global_prefetch_b8)
            if (t + 2 < nT) {
                int pr = tid >> 1;
                if (pr < mLim) __builtin_prefetch(Ab + (long long)pr * lda + (t + 2) * TK, 0, 0);
            }
        }
        // compute tile t: two 32-K WMMA substeps
        const bf16_t* a = sA[cur];
        const bf16_t* b = sB[cur];
#pragma unroll
        for (int ks = 0; ks < TK; ks += 32) {
            v16bf af[2], bfr[4];
            union U { uint4 u[2]; v16bf v; } cv;
#pragma unroll
            for (int i = 0; i < 2; ++i) {
                int row = wm + i * 16 + l16;
                cv.u[0] = *(const uint4*)&a[row * TK + ks + h * 8];        // K 0-7 / 8-15
                cv.u[1] = *(const uint4*)&a[row * TK + ks + 16 + h * 8];   // K 16-23 / 24-31
                af[i] = cv.v;
            }
#pragma unroll
            for (int j = 0; j < 4; ++j) {
                int nn = wn + j * 16 + l16;
                cv.u[0] = *(const uint4*)&b[nn * TK + ks + h * 8];         // K 0-7 / 8-15
                cv.u[1] = *(const uint4*)&b[nn * TK + ks + 16 + h * 8];    // K 16-23 / 24-31
                bfr[j] = cv.v;
            }
#pragma unroll
            for (int i = 0; i < 2; ++i)
#pragma unroll
                for (int j = 0; j < 4; ++j)
                    acc[i][j] = __builtin_amdgcn_wmma_f32_16x16x32_bf16(
                        false, af[i], false, bfr[j], (short)0, acc[i][j], false, false);
        }
        // DMA for t+1 must be complete (and all reads of buffers done) before reuse
        asm volatile("s_wait_asynccnt 0x0" ::: "memory");
        __syncthreads();
    }

    // store (C layout: VGPR r -> M = r + 8*h, N = lane%16)
    long long cb = (long long)blockIdx.z * cBatch;
    for (int i = 0; i < 2; ++i) {
        for (int j = 0; j < 4; ++j) {
            int col = n0 + wn + j * 16 + l16;
            for (int r = 0; r < 8; ++r) {
                int row = m0 + wm + i * 16 + h * 8 + r;
                if (row < M && col < N) {
                    float v = acc[i][j][r] * scale + (bias ? bias[col] : 0.f);
                    if (doRelu) v = fmaxf(v, 0.f);
                    long long idx = cb + (outTrans ? (long long)col * ldc + row
                                                  : (long long)row * ldc + col);
                    if (outBf16) ((bf16_t*)Cout)[idx] = (bf16_t)v;
                    else         ((float*)Cout)[idx]  = v;
                }
            }
        }
    }
}

// ---------------------------------------------------------------------------
// Elementwise / reduction helpers
// ---------------------------------------------------------------------------
__global__ void cvt_bf16_kernel(const float* __restrict__ in, bf16_t* __restrict__ out,
                                long long n) {
    long long i = (long long)blockIdx.x * blockDim.x + threadIdx.x;
    if (i < n) out[i] = (bf16_t)in[i];
}

// out[n*K + k] = (bf16) in[k*N + n]   (W [K,N] -> Wt [N,K])
__global__ void cvt_transpose_kernel(const float* __restrict__ in, bf16_t* __restrict__ out,
                                     int K, int N) {
    long long i = (long long)blockIdx.x * blockDim.x + threadIdx.x;
    if (i >= (long long)K * N) return;
    int n = (int)(i / K), k = (int)(i % K);
    out[i] = (bf16_t)in[(long long)k * N + n];
}

__global__ void colsum_kernel(const float* __restrict__ kf, float* __restrict__ cs,
                              int B, int S, int D) {
    long long i = (long long)blockIdx.x * blockDim.x + threadIdx.x;
    if (i >= (long long)B * D) return;
    int b = (int)(i / D), d = (int)(i % D);
    const float* p = kf + (long long)b * S * D + d;
    float s = 0.f;
    for (int t = 0; t < S; ++t) s += p[(long long)t * D];
    cs[i] = s;
}

// kk[b,s,:] = (colsum + (e-1)*bandsum)/denom_s ; closed form of softmax(mask) @ k
__global__ void kk_kernel(const float* __restrict__ kf, const float* __restrict__ cs,
                          bf16_t* __restrict__ kkb, int B, int S, int D) {
    long long i = (long long)blockIdx.x * blockDim.x + threadIdx.x;
    if (i >= (long long)B * S * D) return;
    int b = (int)(i / ((long long)S * D));
    int rem = (int)(i % ((long long)S * D));
    int s = rem / D, d = rem % D;
    int lo = s - 2 < 0 ? 0 : s - 2;
    int hi = s + 2 >= S ? S - 1 : s + 2;
    const float E = 2.71828182845904523536f;
    float denom = (hi - lo + 1) * E + (float)(S - (hi - lo + 1));
    float band = 0.f;
    const float* p = kf + (long long)b * S * D + d;
    for (int t = lo; t <= hi; ++t) band += p[(long long)t * D];
    kkb[i] = (bf16_t)((cs[(long long)b * D + d] + (E - 1.f) * band) / denom);
}

__global__ __launch_bounds__(256) void softmax_kernel(const float* __restrict__ sc,
                                                      bf16_t* __restrict__ P, int S) {
    __shared__ float red[256];
    long long row = blockIdx.x;
    const float* r = sc + row * (long long)S;
    int tid = threadIdx.x;
    float m = -1e30f;
    for (int i = tid; i < S; i += 256) m = fmaxf(m, r[i]);
    red[tid] = m; __syncthreads();
    for (int s2 = 128; s2 > 0; s2 >>= 1) {
        if (tid < s2) red[tid] = fmaxf(red[tid], red[tid + s2]);
        __syncthreads();
    }
    m = red[0]; __syncthreads();
    float sum = 0.f;
    for (int i = tid; i < S; i += 256) sum += __expf(r[i] - m);
    red[tid] = sum; __syncthreads();
    for (int s2 = 128; s2 > 0; s2 >>= 1) {
        if (tid < s2) red[tid] += red[tid + s2];
        __syncthreads();
    }
    float inv = 1.f / red[0];
    for (int i = tid; i < S; i += 256)
        P[row * (long long)S + i] = (bf16_t)(__expf(r[i] - m) * inv);
}

__global__ void zero_kernel(float* p, int n) {
    int i = blockIdx.x * blockDim.x + threadIdx.x;
    if (i < n) p[i] = 0.f;
}

__global__ __launch_bounds__(256) void reduce_kernel(const float* __restrict__ a,
                                                     const float* __restrict__ b,
                                                     long long n, float* __restrict__ stats) {
    __shared__ float s1[256], s2[256];
    int tid = threadIdx.x;
    float sum = 0.f, sq = 0.f;
    for (long long i = (long long)blockIdx.x * 256 + tid; i < n;
         i += (long long)gridDim.x * 256) {
        float v = a[i] + b[i];
        sum += v; sq += v * v;
    }
    s1[tid] = sum; s2[tid] = sq; __syncthreads();
    for (int s3 = 128; s3 > 0; s3 >>= 1) {
        if (tid < s3) { s1[tid] += s1[tid + s3]; s2[tid] += s2[tid + s3]; }
        __syncthreads();
    }
    if (tid == 0) { atomicAdd(&stats[0], s1[0]); atomicAdd(&stats[1], s2[0]); }
}

__global__ void norm_apply_kernel(const float* __restrict__ a, const float* __restrict__ b,
                                  const float* __restrict__ nw, const float* __restrict__ nb,
                                  void* __restrict__ out, long long n,
                                  const float* __restrict__ stats, int outBf16) {
    long long i = (long long)blockIdx.x * blockDim.x + threadIdx.x;
    if (i >= n) return;
    float mu  = stats[0] / (float)n;
    float var = stats[1] / (float)n - mu * mu;
    float rs  = rsqrtf(var + 1e-6f);
    float v = (a[i] + b[i] - mu) * rs * nw[i] + nb[i];
    if (outBf16) ((bf16_t*)out)[i] = (bf16_t)v;
    else         ((float*)out)[i]  = v;
}

// ---------------------------------------------------------------------------
// Host side
// ---------------------------------------------------------------------------
struct BranchPtrs {
    const float *x, *Wq, *bq, *Wk, *bk, *Wv, *bv, *W1, *b1, *W2, *b2, *nw, *nb;
};

static inline int cdiv_ll(long long a, long long b) { return (int)((a + b - 1) / b); }

static void run_branch(const BranchPtrs& P, float* out, int Bb, int S, int D,
                       char* ws, hipStream_t stream)
{
    const int H = 4 * D;
    const long long BS = (long long)Bb * S;
    const long long nE = BS * D;
    size_t off = 0;
    auto alloc = [&](size_t bytes) {
        void* r = ws + off;
        off = (off + bytes + 255) & ~(size_t)255;
        return r;
    };

    bf16_t* xb  = (bf16_t*)alloc((size_t)nE * 2);
    bf16_t* wqt = (bf16_t*)alloc((size_t)D * D * 2);
    bf16_t* wkt = (bf16_t*)alloc((size_t)D * D * 2);
    bf16_t* wvt = (bf16_t*)alloc((size_t)D * D * 2);
    bf16_t* w1t = (bf16_t*)alloc((size_t)D * H * 2);
    bf16_t* w2t = (bf16_t*)alloc((size_t)H * D * 2);
    bf16_t* qb  = (bf16_t*)alloc((size_t)nE * 2);
    float*  kf  = (float*) alloc((size_t)nE * 4);
    bf16_t* vT  = (bf16_t*)alloc((size_t)nE * 2);   // per-batch [D,S]
    float*  cs  = (float*) alloc((size_t)Bb * D * 4);
    bf16_t* kkb = (bf16_t*)alloc((size_t)nE * 2);
    float*  sc  = (float*) alloc((size_t)Bb * S * S * 4);
    bf16_t* pb  = (bf16_t*)alloc((size_t)Bb * S * S * 2);
    float*  at  = (float*) alloc((size_t)nE * 4);
    bf16_t* y1  = (bf16_t*)alloc((size_t)nE * 2);
    bf16_t* hh  = (bf16_t*)alloc((size_t)BS * H * 2);
    float*  f2  = (float*) alloc((size_t)nE * 4);
    float*  st  = (float*) alloc(16);

    cvt_bf16_kernel<<<cdiv_ll(nE, 256), 256, 0, stream>>>(P.x, xb, nE);
    cvt_transpose_kernel<<<cdiv_ll((long long)D * D, 256), 256, 0, stream>>>(P.Wq, wqt, D, D);
    cvt_transpose_kernel<<<cdiv_ll((long long)D * D, 256), 256, 0, stream>>>(P.Wk, wkt, D, D);
    cvt_transpose_kernel<<<cdiv_ll((long long)D * D, 256), 256, 0, stream>>>(P.Wv, wvt, D, D);
    cvt_transpose_kernel<<<cdiv_ll((long long)D * H, 256), 256, 0, stream>>>(P.W1, w1t, D, H);
    cvt_transpose_kernel<<<cdiv_ll((long long)H * D, 256), 256, 0, stream>>>(P.W2, w2t, H, D);

    auto gemm = [&](const bf16_t* Aa, const bf16_t* Bx, const float* bias, void* C,
                    int M, int N, int K, int lda, int ldb, int ldc,
                    long long ab, long long bb, long long cb,
                    int obf, int otr, int rl, float scl, int batch) {
        dim3 g((N + TN - 1) / TN, (M + TM - 1) / TM, batch);
        gemm_bf16_wmma<<<g, 128, 0, stream>>>(Aa, Bx, bias, C, M, N, K, lda, ldb, ldc,
                                              ab, bb, cb, obf, otr, rl, scl);
    };

    const int M = (int)BS;
    // q, k : batch folded into M; weights transposed [N,K]
    gemm(xb, wqt, P.bq, qb, M, D, D, D, D, D, 0, 0, 0, 1, 0, 0, 1.f, 1);
    gemm(xb, wkt, P.bk, kf, M, D, D, D, D, D, 0, 0, 0, 0, 0, 0, 1.f, 1);
    // v : batched, stored transposed [b][D][S] for use as [N,K] operand later
    gemm(xb, wvt, P.bv, vT, S, D, D, D, D, S,
         (long long)S * D, 0, (long long)S * D, 1, 1, 0, 1.f, Bb);

    // kk = softmax(mask) @ k  (closed form of the dense row-stochastic A)
    colsum_kernel<<<cdiv_ll((long long)Bb * D, 256), 256, 0, stream>>>(kf, cs, Bb, S, D);
    kk_kernel<<<cdiv_ll(nE, 256), 256, 0, stream>>>(kf, cs, kkb, Bb, S, D);

    // scores = (q @ kk^T)/sqrt(D) : B = kk [t][d] already [N,K]
    float iscale = 1.0f / sqrtf((float)D);
    gemm(qb, kkb, nullptr, sc, S, S, D, D, D, S,
         (long long)S * D, (long long)S * D, (long long)S * S, 0, 0, 0, iscale, Bb);

    softmax_kernel<<<Bb * S, 256, 0, stream>>>(sc, pb, S);

    // attn = P @ v : B = vT [d][t] is [N,K]
    gemm(pb, vT, nullptr, at, S, D, S, S, S, D,
         (long long)S * S, (long long)S * D, (long long)S * D, 0, 0, 0, 1.f, Bb);

    // whole_norm(attn + x) -> y1
    zero_kernel<<<1, 32, 0, stream>>>(st, 2);
    reduce_kernel<<<512, 256, 0, stream>>>(at, P.x, nE, st);
    norm_apply_kernel<<<cdiv_ll(nE, 256), 256, 0, stream>>>(at, P.x, P.nw, P.nb, y1, nE, st, 1);

    // FFN
    gemm(y1, w1t, P.b1, hh, M, H, D, D, D, H, 0, 0, 0, 1, 0, 1, 1.f, 1);
    gemm(hh, w2t, P.b2, f2, M, D, H, H, H, D, 0, 0, 0, 0, 0, 0, 1.f, 1);

    // whole_norm(ffn + x) -> final f32
    zero_kernel<<<1, 32, 0, stream>>>(st, 2);
    reduce_kernel<<<512, 256, 0, stream>>>(f2, P.x, nE, st);
    norm_apply_kernel<<<cdiv_ll(nE, 256), 256, 0, stream>>>(f2, P.x, P.nw, P.nb, out, nE, st, 0);
}

extern "C" void kernel_launch(void* const* d_in, const int* in_sizes, int n_in,
                              void* d_out, int out_size, void* d_ws, size_t ws_size,
                              hipStream_t stream)
{
    auto F = [&](int i) { return (const float*)d_in[i]; };
    BranchPtrs tp { F(0), F(2), F(3), F(4), F(5), F(6), F(7),
                    F(8), F(9), F(10), F(11), F(12), F(13) };
    BranchPtrs ip { F(1), F(14), F(15), F(16), F(17), F(18), F(19),
                    F(20), F(21), F(22), F(23), F(24), F(25) };
    float* out = (float*)d_out;
    const int B = 8, St = 512, Dt = 1024, Si = 196, Di = 768;
    run_branch(ip, out + (long long)B * St * Dt, B, Si, Di, (char*)d_ws, stream);
    run_branch(tp, out,                          B, St, Dt, (char*)d_ws, stream);
}
